// TripletLoss_cosine_81982335746653
// MI455X (gfx1250) — matrix operand order repaired and checked by reading, not compiled, not run on hardware
//
#include <hip/hip_runtime.h>
#include <hip/hip_bf16.h>

// Problem constants (match reference)
#define NROWS 8192
#define DIM   1024
#define EPSV  1e-8f

typedef __attribute__((ext_vector_type(16))) _Float16 v16h;
typedef __attribute__((ext_vector_type(8)))  _Float16 v8h;
typedef __attribute__((ext_vector_type(8)))  float    v8f;
typedef __attribute__((ext_vector_type(4)))  float    v4f;
typedef __attribute__((ext_vector_type(4)))  int      v4i;

// ---------------------------------------------------------------------------
// Kernel 1: row-normalize fp32 inputs -> f16 xn (cosine-similarity operands)
// One block (256 threads) per row; 4 elements per thread.
// ---------------------------------------------------------------------------
__global__ __launch_bounds__(256) void tl_normalize_kernel(
    const float* __restrict__ x, _Float16* __restrict__ xnh) {
  int row = blockIdx.x;
  int tid = threadIdx.x;
  const float4 v = *(const float4*)(x + (size_t)row * DIM + tid * 4);
  float ss = v.x * v.x + v.y * v.y + v.z * v.z + v.w * v.w;
  __shared__ float red[256];
  red[tid] = ss;
  __syncthreads();
  for (int s = 128; s > 0; s >>= 1) {
    if (tid < s) red[tid] += red[tid + s];
    __syncthreads();
  }
  float inv = 1.0f / fmaxf(sqrtf(red[0]), EPSV);
  _Float16* o = xnh + (size_t)row * DIM + tid * 4;
  o[0] = (_Float16)(v.x * inv);
  o[1] = (_Float16)(v.y * inv);
  o[2] = (_Float16)(v.z * inv);
  o[3] = (_Float16)(v.w * inv);
}

// Branchless order-statistic maintenance ------------------------------------
__device__ __forceinline__ void push_top3(float x, float& p0, float& p1, float& p2) {
  float t0 = fminf(x, p0);  p0 = fmaxf(x, p0);
  float t1 = fminf(t0, p1); p1 = fmaxf(t0, p1);
  p2 = fmaxf(t1, p2);
}
__device__ __forceinline__ void push_bot6(float x, float& n0, float& n1, float& n2,
                                          float& n3, float& n4, float& n5) {
  float lo;
  lo = fminf(n0, x); x = fmaxf(n0, x); n0 = lo;
  lo = fminf(n1, x); x = fmaxf(n1, x); n1 = lo;
  lo = fminf(n2, x); x = fmaxf(n2, x); n2 = lo;
  lo = fminf(n3, x); x = fmaxf(n3, x); n3 = lo;
  lo = fminf(n4, x); x = fmaxf(n4, x); n4 = lo;
  n5 = fminf(n5, x);
}

// ---------------------------------------------------------------------------
// Kernel 2: fused WMMA GEMM (xn @ xn^T) + per-row top-3 positive / bottom-6
// negative distance selection. Workgroup: 16 rows, 8 waves; wave w handles
// column strips js = w, w+8, ... (each strip = 64 columns, 4 WMMA tiles
// sharing one A fragment, K=1024 in 32 steps of v_wmma_f32_16x16x32_f16).
// ---------------------------------------------------------------------------
__global__ __launch_bounds__(256) void tl_main_kernel(
    const _Float16* __restrict__ xnh, const int* __restrict__ tgt,
    float* __restrict__ rowLoss, float* __restrict__ rowPrec) {
  // per-wave transpose buffer: 16 rows x 64 cols, stride 68 floats (bank pad)
  __shared__ __align__(16) float tileC[8][16][68];
  __shared__ __align__(16) float mergeB[256][9];

  const int lane = threadIdx.x & 31;
  const int wave = threadIdx.x >> 5;
  const int m    = lane & 15;   // row (and B-column) index within tile
  const int h    = lane >> 4;   // half-wave selector
  const int rowBase = blockIdx.x * 16;
  const int grow = rowBase + m;
  const int trow = tgt[grow];

  const float NEG = -__builtin_inff();
  const float POS =  __builtin_inff();
  float p0 = NEG, p1 = NEG, p2 = NEG;                       // 3 largest positive dists
  float n0 = POS, n1 = POS, n2 = POS, n3 = POS, n4 = POS, n5 = POS; // 6 smallest negative

  // A fragment base: lane holds row m; K chunks [k0+16h, +8) and [k0+16h+8, +8)
  const _Float16* Arow = xnh + (size_t)grow * DIM + 16 * h;

  for (int js = wave; js < NROWS / 64; js += 8) {
    const int jbase = js * 64;
    v8f c0{}, c1{}, c2{}, c3{};
    // B fragment base per tile t: lane holds column jbase+16t+m, K chunk [k0+16h, +16)
    const _Float16* B0 = xnh + (size_t)(jbase + m) * DIM + 16 * h;
    const _Float16* B1 = B0 + (size_t)16 * DIM;
    const _Float16* B2 = B1 + (size_t)16 * DIM;
    const _Float16* B3 = B2 + (size_t)16 * DIM;

#pragma unroll 4
    for (int ks = 0; ks < DIM / 32; ++ks) {
      const int k0 = ks * 32;
      v8h a0 = *(const v8h*)(Arow + k0);
      v8h a1 = *(const v8h*)(Arow + k0 + 8);
      v16h a;
#pragma unroll
      for (int i = 0; i < 8; ++i) { a[i] = a0[i]; a[i + 8] = a1[i]; }
      v16h b0 = *(const v16h*)(B0 + k0);
      v16h b1 = *(const v16h*)(B1 + k0);
      v16h b2 = *(const v16h*)(B2 + k0);
      v16h b3 = *(const v16h*)(B3 + k0);
      c0 = __builtin_amdgcn_wmma_f32_16x16x32_f16(false, a, false, b0, (short)0, c0, false, false);
      c1 = __builtin_amdgcn_wmma_f32_16x16x32_f16(false, a, false, b1, (short)0, c1, false, false);
      c2 = __builtin_amdgcn_wmma_f32_16x16x32_f16(false, a, false, b2, (short)0, c2, false, false);
      c3 = __builtin_amdgcn_wmma_f32_16x16x32_f16(false, a, false, b3, (short)0, c3, false, false);
    }

    // transpose C strip through per-wave LDS region: accumulator VGPR r holds
    // D[r+8h][lane%16] (per ISA C/D layout)
#pragma unroll
    for (int r = 0; r < 8; ++r) {
      const int rr = r + 8 * h;
      tileC[wave][rr][ 0 + m] = c0[r];
      tileC[wave][rr][16 + m] = c1[r];
      tileC[wave][rr][32 + m] = c2[r];
      tileC[wave][rr][48 + m] = c3[r];
    }
    __syncthreads();

    // lane (m,h) owns row m, columns [32h, 32h+32) of this strip
#pragma unroll
    for (int q = 0; q < 8; ++q) {
      v4f s  = *(const v4f*)&tileC[wave][m][32 * h + 4 * q];
      v4i tc = *(const v4i*)(tgt + jbase + 32 * h + 4 * q);
#pragma unroll
      for (int e = 0; e < 4; ++e) {
        const float d   = 1.0f - s[e];
        const bool  pos = (tc[e] == trow);
        const float dp  = pos ? d : NEG;   // -inf never enters top-3
        const float dn  = pos ? POS : d;   // +inf never enters bottom-6
        push_top3(dp, p0, p1, p2);
        push_bot6(dn, n0, n1, n2, n3, n4, n5);
      }
    }
    __syncthreads();
  }

  // merge 16 partial lists (8 waves x 2 halves) per row
  mergeB[threadIdx.x][0] = p0; mergeB[threadIdx.x][1] = p1; mergeB[threadIdx.x][2] = p2;
  mergeB[threadIdx.x][3] = n0; mergeB[threadIdx.x][4] = n1; mergeB[threadIdx.x][5] = n2;
  mergeB[threadIdx.x][6] = n3; mergeB[threadIdx.x][7] = n4; mergeB[threadIdx.x][8] = n5;
  __syncthreads();

  if (threadIdx.x < 16) {
    const int mm = threadIdx.x;
    float P0 = NEG, P1 = NEG, P2 = NEG;
    float N0 = POS, N1 = POS, N2 = POS, N3 = POS, N4 = POS, N5 = POS;
    for (int w = 0; w < 8; ++w) {
      for (int hh = 0; hh < 2; ++hh) {
        const float* q = &mergeB[w * 32 + hh * 16 + mm][0];
        push_top3(q[0], P0, P1, P2);
        push_top3(q[1], P0, P1, P2);
        push_top3(q[2], P0, P1, P2);
        push_bot6(q[3], N0, N1, N2, N3, N4, N5);
        push_bot6(q[4], N0, N1, N2, N3, N4, N5);
        push_bot6(q[5], N0, N1, N2, N3, N4, N5);
        push_bot6(q[6], N0, N1, N2, N3, N4, N5);
        push_bot6(q[7], N0, N1, N2, N3, N4, N5);
        push_bot6(q[8], N0, N1, N2, N3, N4, N5);
      }
    }
    const float ap = P2;  // 3rd-largest positive distance
    const float an = N5;  // 6th-smallest negative distance
    rowLoss[rowBase + mm] = fmaxf(0.0f, ap - an);   // relu(margin-(an-ap)), margin=0
    rowPrec[rowBase + mm] = (an > ap) ? 1.0f : 0.0f;
  }
}

// ---------------------------------------------------------------------------
// Kernel 3: deterministic fixed-order reduction of per-row loss / precision.
// ---------------------------------------------------------------------------
__global__ __launch_bounds__(256) void tl_reduce_kernel(
    const float* __restrict__ rowLoss, const float* __restrict__ rowPrec,
    float* __restrict__ out) {
  __shared__ float sl[256];
  __shared__ float sp[256];
  const int tid = threadIdx.x;
  float a = 0.0f, b = 0.0f;
  for (int i = tid; i < NROWS; i += 256) { a += rowLoss[i]; b += rowPrec[i]; }
  sl[tid] = a; sp[tid] = b;
  __syncthreads();
  for (int s = 128; s > 0; s >>= 1) {
    if (tid < s) { sl[tid] += sl[tid + s]; sp[tid] += sp[tid + s]; }
    __syncthreads();
  }
  if (tid == 0) {
    out[0] = sl[0] / (float)NROWS;  // mean loss
    out[1] = sp[0];                 // prec count
  }
}

extern "C" void kernel_launch(void* const* d_in, const int* in_sizes, int n_in,
                              void* d_out, int out_size, void* d_ws, size_t ws_size,
                              hipStream_t stream) {
  (void)in_sizes; (void)n_in; (void)out_size; (void)ws_size;
  const float* x   = (const float*)d_in[0];
  const int*   tgt = (const int*)d_in[1];
  float*       out = (float*)d_out;

  // workspace layout: [xn f16: 8192*1024*2 B][rowLoss: 8192 f32][rowPrec: 8192 f32]
  char* ws = (char*)d_ws;
  _Float16* xnh     = (_Float16*)ws;
  float*    rowLoss = (float*)(ws + (size_t)NROWS * DIM * sizeof(_Float16));
  float*    rowPrec = rowLoss + NROWS;

  tl_normalize_kernel<<<NROWS, 256, 0, stream>>>(x, xnh);
  tl_main_kernel<<<NROWS / 16, 256, 0, stream>>>(xnh, tgt, rowLoss, rowPrec);
  tl_reduce_kernel<<<1, 256, 0, stream>>>(rowLoss, rowPrec, out);
}
